// SelfAttentionHead_8143257993621
// MI455X (gfx1250) — matrix operand order repaired and checked by reading, not compiled, not run on hardware
//
#include <hip/hip_runtime.h>
#include <hip/hip_bf16.h>

// ---- problem constants (match reference) ----
#define Bsz 8
#define Tsz 2048
#define Csz 1024
#define Hsz 64

typedef __attribute__((ext_vector_type(16))) _Float16 v16h;
typedef __attribute__((ext_vector_type(8)))  _Float16 v8h;
typedef __attribute__((ext_vector_type(8)))  float    v8f;
typedef __attribute__((ext_vector_type(4)))  float    f32x4;

union V16 { v16h v; v8h h[2]; };

__device__ inline v8f wmma_f16(v16h a, v16h b, v8f c) {
  // D = A(16x32 f16) * B(32x16 f16) + C(16x16 f32)
  return __builtin_amdgcn_wmma_f32_16x16x32_f16(
      /*neg_a=*/false, a, /*neg_b=*/false, b,
      /*c_mod=*/(short)0, c, /*reuse_a=*/false, /*reuse_b=*/false);
}

// =====================================================================
// Kernel 1: fused Q/K/V projection, fp32 x -> fp16 Q (pre-scaled), K, V^T
// 256 threads = 8 waves per block (fills a WGP under wave32), each wave
// computes 16 tokens x 64 head-dims for all three projections. Weight
// chunk (32 x 64, transposed) staged in LDS each K-step; halving block
// count halves redundant W reads (W stays L2-resident in 192MB L2).
// =====================================================================
__global__ __launch_bounds__(256)
void qkv_project_kernel(const float* __restrict__ x,
                        const float* __restrict__ Wq,
                        const float* __restrict__ Wk,
                        const float* __restrict__ Wv,
                        _Float16* __restrict__ Qh,   // [B][T][H], scaled 1/32
                        _Float16* __restrict__ Kh,   // [B][T][H]
                        _Float16* __restrict__ Vt)   // [B][H][T]
{
  __shared__ _Float16 ldsw[3][Hsz][32];   // 12 KB: W chunk, transposed [h][kk]

  const int tid  = threadIdx.x;
  const int lane = tid & 31;
  const int wave = tid >> 5;
  const int tile = blockIdx.x * 8 + wave;     // 16-token tile id, 1024 total
  const int b    = tile >> 7;                 // 128 tiles per batch
  const int t0   = (tile & 127) << 4;
  const int row  = lane & 15;
  const int hi   = lane >> 4;                 // 0: lanes 0-15, 1: lanes 16-31

  const float* xrow = x + (size_t)(b * Tsz + t0 + row) * Csz;
  const float* Wp[3] = {Wq, Wk, Wv};

  v8f acc[3][4];
  #pragma unroll
  for (int m = 0; m < 3; ++m)
    #pragma unroll
    for (int nt = 0; nt < 4; ++nt)
      #pragma unroll
      for (int j = 0; j < 8; ++j) acc[m][nt][j] = 0.0f;

  for (int c0 = 0; c0 < Csz; c0 += 32) {
    __syncthreads();
    // ---- stage weight chunk [32 x 64] -> LDS transposed [64][32], f16 ----
    #pragma unroll
    for (int m = 0; m < 3; ++m) {
      const float* W = Wp[m];
      for (int i = tid; i < 32 * Hsz; i += 256) {
        int kk = i >> 6;       // 0..31  (C-chunk row)
        int h  = i & 63;       // 0..63  (coalesced over h)
        ldsw[m][h][kk] = (_Float16)W[(c0 + kk) * Hsz + h];
      }
    }
    __syncthreads();

    // ---- A fragment: x tile 16 tokens x 32 channels, fp32 -> f16 ----
    V16 a;
    {
      const int o1 = hi ? 8 : 0;
      f32x4 f0 = *(const f32x4*)(xrow + c0 + o1);
      f32x4 f1 = *(const f32x4*)(xrow + c0 + o1 + 4);
      f32x4 f2 = *(const f32x4*)(xrow + c0 + o1 + 16);
      f32x4 f3 = *(const f32x4*)(xrow + c0 + o1 + 20);
      v8h lo, hi8;
      #pragma unroll
      for (int j = 0; j < 4; ++j) {
        lo[j]      = (_Float16)f0[j];
        lo[j + 4]  = (_Float16)f1[j];
        hi8[j]     = (_Float16)f2[j];
        hi8[j + 4] = (_Float16)f3[j];
      }
      a.h[0] = lo; a.h[1] = hi8;
    }

    // ---- 12 WMMAs: 3 projections x 4 output n-tiles ----
    #pragma unroll
    for (int m = 0; m < 3; ++m) {
      #pragma unroll
      for (int nt = 0; nt < 4; ++nt) {
        V16 bf;
        const v8h* p = (const v8h*)&ldsw[m][nt * 16 + row][hi ? 8 : 0];
        bf.h[0] = p[0];
        bf.h[1] = p[2];
        acc[m][nt] = wmma_f16(a.v, bf.v, acc[m][nt]);
      }
    }
  }

  // ---- store: Q (scaled by C^-0.5 = 1/32), K row-major, V transposed ----
  const float qscale = 0.03125f;
  #pragma unroll
  for (int nt = 0; nt < 4; ++nt) {
    #pragma unroll
    for (int r = 0; r < 8; ++r) {
      const int mr = r + 8 * hi;
      const int t  = t0 + mr;
      const int h  = nt * 16 + row;
      const size_t idx = (size_t)(b * Tsz + t) * Hsz + h;
      Qh[idx] = (_Float16)(acc[0][nt][r] * qscale);
      Kh[idx] = (_Float16)(acc[1][nt][r]);
      Vt[(size_t)(b * Hsz + h) * Tsz + t] = (_Float16)(acc[2][nt][r]);
    }
  }
}

// =====================================================================
// Kernel 2: causal flash attention. 4 waves/block, one 16-query tile
// per wave (adjacent tiles). K/V 32-key blocks staged cooperatively in
// LDS once per block (4x cut in L2 traffic); key loop is block-uniform
// (bounded by the block's max q0) so __syncthreads is legal -- waves
// past their causal frontier are fully masked (alpha=1, p=0).
// =====================================================================
__global__ __launch_bounds__(128)
void attn_kernel(const _Float16* __restrict__ Qh,
                 const _Float16* __restrict__ Kh,
                 const _Float16* __restrict__ Vt,
                 float* __restrict__ out)      // [B][T][H] fp32
{
  __shared__ _Float16 kblk[32][Hsz];          // [key][h]   4 KB
  __shared__ _Float16 vblk[Hsz][32];          // [h][key]   4 KB
  __shared__ _Float16 pstage[4][16][32];      // per-wave P 4 KB

  const int tid  = threadIdx.x;
  const int lane = tid & 31;
  const int wave = tid >> 5;
  const int tile = blockIdx.x * 4 + wave;
  const int b    = tile >> 7;                 // whole block is in one batch
  const int q0   = (tile & 127) << 4;
  const int row  = lane & 15;
  const int hi   = lane >> 4;
  const int o1   = hi ? 8 : 0;

  // ---- Q A-fragments for K-dim H=64: two 16x32 fragments ----
  V16 qa[2];
  {
    const _Float16* qrow = Qh + (size_t)(b * Tsz + q0 + row) * Hsz;
    #pragma unroll
    for (int hc = 0; hc < 2; ++hc) {
      qa[hc].h[0] = *(const v8h*)(qrow + hc * 32 + o1);
      qa[hc].h[1] = *(const v8h*)(qrow + hc * 32 + o1 + 16);
    }
  }

  v8f oacc[4];
  float mrow[8], lrow[8];
  #pragma unroll
  for (int nt = 0; nt < 4; ++nt)
    #pragma unroll
    for (int j = 0; j < 8; ++j) oacc[nt][j] = 0.0f;
  #pragma unroll
  for (int r = 0; r < 8; ++r) { mrow[r] = -1e30f; lrow[r] = 0.0f; }

  // block-uniform causal bound: last wave in block has the largest q0
  const int q0max = (((blockIdx.x * 4 + 3) & 127) << 4);
  const int nkb   = (q0max + 15) / 32 + 1;

  // per-thread staging roles
  const int kst = tid >> 2;            // 0..31 : key row for K staging
  const int khs = (tid & 3) << 4;      // 0/16/32/48 : h segment
  const int vsh = tid >> 1;            // 0..63 : h row for V staging
  const int vts = (tid & 1) << 4;      // 0/16 : key segment

  for (int kb = 0; kb < nkb; ++kb) {
    const int kt0 = kb * 32;

    __syncthreads();   // previous iteration's LDS reads are done
    // ---- cooperative staging: K block [32][64], V block [64][32] ----
    {
      const v8h* ks = (const v8h*)(Kh + (size_t)(b * Tsz + kt0 + kst) * Hsz + khs);
      v8h* kd = (v8h*)&kblk[kst][khs];
      kd[0] = ks[0]; kd[1] = ks[1];

      const v8h* vs = (const v8h*)(Vt + (size_t)(b * Hsz + vsh) * Tsz + kt0 + vts);
      v8h* vd = (v8h*)&vblk[vsh][vts];
      vd[0] = vs[0]; vd[1] = vs[1];

      if (kt0 + 32 < Tsz) {  // prefetch next key block (global_prefetch_b8)
        __builtin_prefetch(Kh + (size_t)(b * Tsz + kt0 + 32 + kst) * Hsz + khs, 0, 1);
        __builtin_prefetch(Vt + (size_t)(b * Hsz + vsh) * Tsz + kt0 + 32 + vts, 0, 1);
      }
    }
    __syncthreads();

    // ---- S = Q K^T for two 16-key sub-tiles (fragments from LDS) ----
    v8f s[2];
    #pragma unroll
    for (int nt = 0; nt < 2; ++nt) {
      #pragma unroll
      for (int j = 0; j < 8; ++j) s[nt][j] = 0.0f;
      #pragma unroll
      for (int hc = 0; hc < 2; ++hc) {
        V16 kf;
        const v8h* kp = (const v8h*)&kblk[nt * 16 + row][hc * 32 + (hi ? 8 : 0)];
        kf.h[0] = kp[0];
        kf.h[1] = kp[2];
        s[nt] = wmma_f16(qa[hc].v, kf.v, s[nt]);
      }
    }

    // ---- causal mask + online softmax (row-wise over 16-lane groups) ----
    #pragma unroll
    for (int r = 0; r < 8; ++r) {
      const int q  = q0 + r + 8 * hi;
      const int k0 = kt0 + row;
      const int k1 = kt0 + 16 + row;
      float s0 = (k0 <= q) ? s[0][r] : -1e30f;
      float s1 = (k1 <= q) ? s[1][r] : -1e30f;

      float t = fmaxf(s0, s1);
      #pragma unroll
      for (int off = 8; off >= 1; off >>= 1) t = fmaxf(t, __shfl_xor(t, off));
      const float mn    = fmaxf(mrow[r], t);
      const float alpha = __expf(mrow[r] - mn);
      mrow[r] = mn;

      const float p0 = __expf(s0 - mn);
      const float p1 = __expf(s1 - mn);
      float rs = p0 + p1;
      #pragma unroll
      for (int off = 8; off >= 1; off >>= 1) rs += __shfl_xor(rs, off);
      lrow[r] = lrow[r] * alpha + rs;

      #pragma unroll
      for (int nt = 0; nt < 4; ++nt) oacc[nt][r] *= alpha;

      // stage P (C-layout -> LDS row-major 16x32, f16)
      pstage[wave][r + 8 * hi][row]      = (_Float16)p0;
      pstage[wave][r + 8 * hi][16 + row] = (_Float16)p1;
    }

    // wave-local LDS RAW: wait for ds stores before cross-lane reads
    __builtin_amdgcn_wave_barrier();
    asm volatile("s_wait_dscnt 0" ::: "memory");

    // ---- reload P as 16x32 A-fragment ----
    V16 pa;
    {
      const v8h* pr = (const v8h*)&pstage[wave][row][o1];
      pa.h[0] = pr[0];   // halves o1 .. o1+7
      pa.h[1] = pr[2];   // halves o1+16 .. o1+23
    }
    __builtin_amdgcn_wave_barrier();

    // ---- O += P V : 4 output h-tiles (V fragments from LDS) ----
    #pragma unroll
    for (int nt = 0; nt < 4; ++nt) {
      V16 vf;
      const v8h* vp = (const v8h*)&vblk[nt * 16 + row][hi ? 8 : 0];
      vf.h[0] = vp[0];
      vf.h[1] = vp[2];
      oacc[nt] = wmma_f16(pa.v, vf.v, oacc[nt]);
    }
  }

  // ---- normalize and store fp32 output ----
  #pragma unroll
  for (int nt = 0; nt < 4; ++nt) {
    #pragma unroll
    for (int r = 0; r < 8; ++r) {
      const int t = q0 + r + 8 * hi;
      const int h = nt * 16 + row;
      out[(size_t)(b * Tsz + t) * Hsz + h] = oacc[nt][r] / lrow[r];
    }
  }
}

extern "C" void kernel_launch(void* const* d_in, const int* in_sizes, int n_in,
                              void* d_out, int out_size, void* d_ws, size_t ws_size,
                              hipStream_t stream) {
  const float* x  = (const float*)d_in[0];
  const float* Wq = (const float*)d_in[1];
  const float* Wk = (const float*)d_in[2];
  const float* Wv = (const float*)d_in[3];
  float* out = (float*)d_out;

  // workspace: three f16 tensors of B*T*H = 1M halves each (6 MB total)
  _Float16* Qh = (_Float16*)d_ws;
  _Float16* Kh = Qh + (size_t)Bsz * Tsz * Hsz;
  _Float16* Vt = Kh + (size_t)Bsz * Tsz * Hsz;

  // projection: 128 blocks x 8 waves x 16 tokens = B*T = 16384 tokens
  qkv_project_kernel<<<128, 256, 0, stream>>>(x, Wq, Wk, Wv, Qh, Kh, Vt);
  // attention: 256 blocks x 4 waves x 16 queries = B*T
  attn_kernel<<<256, 128, 0, stream>>>(Qh, Kh, Vt, out);
}
